// WindowAttnBlock_26929444946409
// MI455X (gfx1250) — matrix operand order repaired and checked by reading, not compiled, not run on hardware
//
#include <hip/hip_runtime.h>
#include <hip/hip_bf16.h>
#include <stdint.h>

// ---------------- problem constants ----------------
#define Bsz   32
#define Cdim  256
#define Hdim  56
#define Wdim  56
#define NHead 8
#define HD    32          // head dim
#define Ntok  49          // tokens per window
#define Lseq  3136        // H*W
#define Mrows 100352      // B * L  (== B*nW*Ntok)
#define NWIN  2048        // B * 64 windows
#define QKV_STRIDE 25690112ull   // Mrows * Cdim elements per q/k/v plane
#define SCALE_Q 0.17677669529663687f  // 1/sqrt(32)

typedef __bf16 bf16;
typedef __attribute__((ext_vector_type(16))) __bf16 v16bf;
typedef __attribute__((ext_vector_type(8)))  float  v8f;
typedef __attribute__((ext_vector_type(4)))  unsigned u32x4;
typedef __attribute__((ext_vector_type(8)))  int      i32x8;
typedef __attribute__((ext_vector_type(4)))  int      i32x4;

union FragBF { v16bf v; bf16 e[16]; u32x4 q2[2]; };
union FragF  { v8f   v; float e[8];  };
union PairBF { unsigned u; bf16 h[2]; };
union OctBF  { u32x4 u4; bf16 h[8]; };

__device__ __forceinline__ int reg56(int t) { return (t < 49) ? 0 : ((t < 53) ? 1 : 2); }

// ---------------- Tensor Data Mover (TDM) support ----------------
#if defined(__has_builtin)
#if __has_builtin(__builtin_amdgcn_tensor_load_to_lds) && \
    __has_builtin(__builtin_amdgcn_s_wait_tensorcnt)
#define USE_TDM 1
#endif
#endif
#ifndef USE_TDM
#define USE_TDM 0
#endif

#if USE_TDM
// 2-D row-major tile load, data_size=8B units. Rows >= tensor_rows read as 0.
__device__ __forceinline__ void tdm_load_tile(const void* lds_dst, const void* gsrc,
                                              unsigned row_units,    // tile width, 8B units
                                              unsigned tile_rows,    // tile height
                                              unsigned stride_units, // tensor row stride, 8B units
                                              unsigned tensor_rows)  // valid rows (OOB -> 0)
{
    unsigned long long ga = (unsigned long long)gsrc;
    unsigned lds = (unsigned)(unsigned long long)lds_dst;   // low 32b of flat = LDS offset
    u32x4 g0 = { 1u,                                  // count=1, user mode
                 lds,                                 // lds_addr
                 (unsigned)ga,                        // global_addr[31:0]
                 (unsigned)(ga >> 32) | (2u << 30) }; // global_addr[56:32] | type=2
    unsigned td0 = stride_units;                      // tensor width = stride (no x-OOB)
    unsigned td1 = tensor_rows;
    i32x8 g1 = { (int)(3u << 16),                               // data_size=3 (8B)
                 (int)((td0 & 0xffffu) << 16),                  // tensor_dim0[15:0]
                 (int)((td0 >> 16) | ((td1 & 0xffffu) << 16)),  // td0[31:16] | td1[15:0]
                 (int)((td1 >> 16) | (row_units << 16)),        // td1[31:16] | tile_dim0
                 (int)tile_rows,                                // tile_dim1 (tile_dim2=0)
                 (int)stride_units,                             // tensor_dim0_stride[31:0]
                 0, 0 };
    i32x4 gz = { 0, 0, 0, 0 };
#if __clang_major__ >= 23
    i32x8 gz8 = { 0, 0, 0, 0, 0, 0, 0, 0 };
    __builtin_amdgcn_tensor_load_to_lds(g0, g1, gz, gz, gz8, 0);
#else
    __builtin_amdgcn_tensor_load_to_lds(g0, g1, gz, gz, 0);
#endif
}
#endif

// =====================================================================
// Kernel 1: LayerNorm1 + cyclic shift + window partition.
//  x (B,C,H,W) f32 -> xs (B,L,C) f32 residual copy, xw (nWin*49, C) bf16
// =====================================================================
__global__ __launch_bounds__(64) void ln1_part_kernel(
    const float* __restrict__ x, const float* __restrict__ nw,
    const float* __restrict__ nb, float* __restrict__ xs, bf16* __restrict__ xw)
{
    int t = threadIdx.x;
    if (t >= Wdim) return;
    int b = blockIdx.x / Hdim;
    int h = blockIdx.x - b * Hdim;
    const float* xp = x + (size_t)b * Cdim * Lseq + h * Wdim + t;

    float s = 0.f, sq = 0.f;
    for (int c = 0; c < Cdim; ++c) {
        float v = xp[(size_t)c * Lseq];
        s += v; sq += v * v;
    }
    float mu   = s * (1.f / Cdim);
    float rstd = rsqrtf(sq * (1.f / Cdim) - mu * mu + 1e-5f);

    int hs = h + 53; if (hs >= 56) hs -= 56;   // roll by -3
    int ws = t + 53; if (ws >= 56) ws -= 56;
    int wh = hs / 7, ii = hs - 7 * wh;
    int ww = ws / 7, jj = ws - 7 * ww;
    size_t mrow = ((size_t)b * 64 + wh * 8 + ww) * Ntok + ii * 7 + jj;

    float* xsp = xs + ((size_t)b * Lseq + h * Wdim + t) * Cdim;
    bf16*  xwp = xw + mrow * Cdim;
    for (int c = 0; c < Cdim; ++c) {
        float v = xp[(size_t)c * Lseq];
        xsp[c] = v;
        xwp[c] = (bf16)((v - mu) * rstd * nw[c] + nb[c]);
    }
}

// =====================================================================
// bf16 WMMA GEMM, D = A(MxK) * W(NxK)^T, block tile 128x64, 8 waves.
// A tile staged via TDM (tensor_load_to_lds) when available.
// B tile staged transposed so fragments are contiguous ds_load_b128.
// EPI: 0=qkv scatter, 1=proj+unshift residual, 2=fc1 relu, 3=fc2 out
// =====================================================================
template <int EPI>
__global__ __launch_bounds__(256) void gemm_bf16_kernel(
    const bf16* __restrict__ A, const bf16* __restrict__ W, int K,
    const float* __restrict__ bias0, const float* __restrict__ bias1,
    bf16* __restrict__ obf, float* __restrict__ of32)
{
    __shared__ alignas(16) bf16 As[128 * 32];   // [m][k]
    __shared__ alignas(16) bf16 Bt[32 * 64];    // [n][k]

    const int tid  = threadIdx.x;
    const int wave = tid >> 5, lane = tid & 31;
    const int wm = wave >> 1, wn = wave & 1;
    const int m0 = blockIdx.x * 128, n0 = blockIdx.y * 64;
    const int Kq = K >> 3;                      // 8-elem groups per row
    const u32x4* Wg = (const u32x4*)W;
#if !USE_TDM
    const u32x4* Ag = (const u32x4*)A;
#endif

    FragF acc[2][2];
#pragma unroll
    for (int ti = 0; ti < 2; ++ti)
#pragma unroll
        for (int tj = 0; tj < 2; ++tj)
#pragma unroll
            for (int r = 0; r < 8; ++r) acc[ti][tj].e[r] = 0.f;

    const int rlo = lane & 15;
    const int klo = (lane < 16) ? 0 : 8;

    for (int k0 = 0; k0 < K; k0 += 32) {
#if USE_TDM
        if (wave == 0)  // one TDM op moves the whole 128x32 bf16 A tile
            tdm_load_tile(As, A + (size_t)m0 * K + k0,
                          8u, 128u, (unsigned)(K >> 2), 1u << 20);
#else
#pragma unroll
        for (int i = 0; i < 2; ++i) {
            int f = tid * 2 + i, ar = f >> 2, aq = f & 3;
            ((u32x4*)As)[ar * 4 + aq] = Ag[(size_t)(m0 + ar) * Kq + (k0 >> 3) + aq];
        }
#endif
        {   // stage W tile: Bt[n][k] (W row order)
            int br = tid >> 2, bq = tid & 3;
            OctBF w;
            w.u4 = Wg[(size_t)(n0 + br) * Kq + (k0 >> 3) + bq];
#pragma unroll
            for (int t = 0; t < 8; ++t)
                Bt[br * 32 + bq * 8 + t] = w.h[t];
        }
        if (k0 + 32 < K) {  // global_prefetch_b8 for next W chunk
            __builtin_prefetch(&Wg[(size_t)(n0 + (tid >> 2)) * Kq + ((k0 + 32) >> 3)], 0, 0);
        }
#if USE_TDM
        if (wave == 0) __builtin_amdgcn_s_wait_tensorcnt((short)0);
#endif
        __syncthreads();

        FragBF a[2], b[2];
#pragma unroll
        for (int t = 0; t < 2; ++t) {
            int row = wm * 32 + t * 16 + rlo;
            a[t].q2[0] = *(const u32x4*)(As + row * 32 + klo);       // K klo..klo+7
            a[t].q2[1] = *(const u32x4*)(As + row * 32 + 16 + klo);  // K 16+klo..
            int nbase = wn * 32 + t * 16;
            b[t].q2[0] = *(const u32x4*)(Bt + (nbase + rlo) * 32 + klo);       // K klo..klo+7
            b[t].q2[1] = *(const u32x4*)(Bt + (nbase + rlo) * 32 + 16 + klo);
        }
#pragma unroll
        for (int ti = 0; ti < 2; ++ti)
#pragma unroll
            for (int tj = 0; tj < 2; ++tj)
                acc[ti][tj].v = __builtin_amdgcn_wmma_f32_16x16x32_bf16(
                    false, a[ti].v, false, b[tj].v, (short)0, acc[ti][tj].v, false, false);
        __syncthreads();
    }

    // ---------------- fused epilogues ----------------
    const int rofs = (lane >= 16) ? 8 : 0;
#pragma unroll
    for (int ti = 0; ti < 2; ++ti) {
#pragma unroll
        for (int tj = 0; tj < 2; ++tj) {
#pragma unroll
            for (int r = 0; r < 8; ++r) {
                int m = m0 + wm * 32 + ti * 16 + rofs + r;
                int n = n0 + wn * 32 + tj * 16 + rlo;
                float val = acc[ti][tj].e[r];
                if (EPI == 0) {          // qkv: bias, scale q, scatter per (win,head)
                    int which = n >> 8, rem = n & 255;
                    int head = rem >> 5, d = rem & 31;
                    int win = m / 49, tok = m - win * 49;
                    float bia = (which == 0) ? bias0[rem] : ((which == 2) ? bias1[rem] : 0.f);
                    float v = val + bia;
                    if (which == 0) v *= SCALE_Q;
                    obf[(size_t)which * QKV_STRIDE +
                        (((size_t)(win * 8 + head)) * Ntok + tok) * HD + d] = (bf16)v;
                } else if (EPI == 1) {   // proj: window reverse + unshift + residual
                    float v = val + bias0[n];
                    int win = m / 49, tok = m - win * 49;
                    int b = win >> 6, wloc = win & 63;
                    int wh = wloc >> 3, ww = wloc & 7;
                    int ii = tok / 7, jj = tok - 7 * ii;
                    int h = wh * 7 + ii + 3; if (h >= 56) h -= 56;
                    int w = ww * 7 + jj + 3; if (w >= 56) w -= 56;
                    of32[(((size_t)b * Lseq + h * Wdim + w) * Cdim) + n] += v;
                } else if (EPI == 2) {   // fc1 + relu -> bf16
                    obf[(size_t)m * 1024 + n] = (bf16)fmaxf(val + bias0[n], 0.f);
                } else {                 // fc2 + residual, write NCHW f32
                    float v = val + bias0[n] + bias1[(size_t)m * Cdim + n];
                    int b = m / Lseq, hw = m - b * Lseq;
                    of32[((size_t)b * Cdim + n) * Lseq + hw] = v;
                }
            }
        }
    }
}

// =====================================================================
// Window attention: one block per (window, head), 128 threads.
// q/v staged row-major (TDM zero-pads rows>=49), k staged row-major [m][d].
// S = qk^T (4 wmma), softmax w/ analytic bias+mask, O = P*V (4 wmma).
// =====================================================================
__global__ __launch_bounds__(128) void attn_kernel(
    const bf16* __restrict__ qkv, const float* __restrict__ rpb, bf16* __restrict__ out)
{
    __shared__ alignas(16) bf16 qs[64 * 32];   // [m][d]
    __shared__ alignas(16) bf16 kt[32 * 64];   // [m][d] row-major
    __shared__ alignas(16) bf16 vs[64 * 32];   // [m][d]
    __shared__ float Ss[64 * 64];
    __shared__ alignas(16) bf16 Ps[64 * 64];

    const int tid  = threadIdx.x;
    const int win  = blockIdx.x >> 3;
    const int head = blockIdx.x & 7;
    const size_t base = ((size_t)(win * 8 + head)) * Ntok * HD;

    const bf16* qb = qkv + base;
    const bf16* kb = qkv + QKV_STRIDE + base;
    const bf16* vb = qkv + 2 * QKV_STRIDE + base;
    const unsigned* kg = (const unsigned*)kb;
    const int wave = tid >> 5, lane = tid & 31;

#if USE_TDM
    if (wave == 0) {   // rows 49..63 auto-zero via tensor_dim1 = 49 OOB rule
        tdm_load_tile(qs, qb, 8u, 64u, 8u, (unsigned)Ntok);
        tdm_load_tile(vs, vb, 8u, 64u, 8u, (unsigned)Ntok);
    }
#else
    const unsigned* qg = (const unsigned*)qb;
    const unsigned* vg = (const unsigned*)vb;
#endif
    for (int idx = tid; idx < 64 * 16; idx += 128) {   // 2 bf16 per dword
        int row = idx >> 4, dp = idx & 15;
        PairBF ku; ku.u = (row < Ntok) ? kg[idx] : 0u;
        kt[row * 32 + 2 * dp]     = ku.h[0];
        kt[row * 32 + 2 * dp + 1] = ku.h[1];
#if !USE_TDM
        ((unsigned*)qs)[idx] = (row < Ntok) ? qg[idx] : 0u;
        ((unsigned*)vs)[idx] = (row < Ntok) ? vg[idx] : 0u;
#endif
    }
#if USE_TDM
    if (wave == 0) __builtin_amdgcn_s_wait_tensorcnt((short)0);
#endif
    __syncthreads();

    const int m0 = wave * 16;
    const int rlo = lane & 15;
    const int klo = (lane < 16) ? 0 : 8;
    const int rofs = (lane >= 16) ? 8 : 0;

    // ---- S = q k^T ----
    FragBF a;
    a.q2[0] = *(const u32x4*)(qs + (m0 + rlo) * 32 + klo);
    a.q2[1] = *(const u32x4*)(qs + (m0 + rlo) * 32 + 16 + klo);
#pragma unroll
    for (int nt = 0; nt < 4; ++nt) {
        FragBF b;
        b.q2[0] = *(const u32x4*)(kt + (nt * 16 + rlo) * 32 + klo);
        b.q2[1] = *(const u32x4*)(kt + (nt * 16 + rlo) * 32 + 16 + klo);
        FragF c;
#pragma unroll
        for (int r = 0; r < 8; ++r) c.e[r] = 0.f;
        c.v = __builtin_amdgcn_wmma_f32_16x16x32_bf16(false, a.v, false, b.v,
                                                      (short)0, c.v, false, false);
#pragma unroll
        for (int r = 0; r < 8; ++r)
            Ss[(m0 + rofs + r) * 64 + nt * 16 + rlo] = c.e[r];
    }
    __syncthreads();

    // ---- softmax with analytic rel-pos bias + shift mask ----
    const int wloc = win & 63, whh = wloc >> 3, www = wloc & 7;
    if (tid < 64) {
        int row = tid;
        if (row < Ntok) {
            int i1 = row / 7, j1 = row - 7 * i1;
            int rn = reg56(whh * 7 + i1) * 3 + reg56(www * 7 + j1);
            float mx = -1e30f;
            for (int m = 0; m < Ntok; ++m) {
                int i2 = m / 7, j2 = m - 7 * i2;
                float bia = rpb[((i1 - i2 + 6) * 13 + (j1 - j2 + 6)) * NHead + head];
                int rm = reg56(whh * 7 + i2) * 3 + reg56(www * 7 + j2);
                float v = Ss[row * 64 + m] + bia + ((rm != rn) ? -100.f : 0.f);
                Ss[row * 64 + m] = v;
                mx = fmaxf(mx, v);
            }
            float sum = 0.f;
            for (int m = 0; m < Ntok; ++m) {
                float e = __expf(Ss[row * 64 + m] - mx);
                Ss[row * 64 + m] = e;
                sum += e;
            }
            float inv = 1.f / sum;
            for (int m = 0; m < 64; ++m)
                Ps[row * 64 + m] = (m < Ntok) ? (bf16)(Ss[row * 64 + m] * inv) : (bf16)0.f;
        } else {
            for (int m = 0; m < 64; ++m) Ps[row * 64 + m] = (bf16)0.f;
        }
    }
    __syncthreads();

    // ---- O = P * V  (K = 64 in two 32-chunks; V rows >= 49 are zero) ----
#pragma unroll
    for (int jt = 0; jt < 2; ++jt) {
        FragF c;
#pragma unroll
        for (int r = 0; r < 8; ++r) c.e[r] = 0.f;
#pragma unroll
        for (int kc = 0; kc < 2; ++kc) {
            FragBF a2, b2;
            a2.q2[0] = *(const u32x4*)(Ps + (m0 + rlo) * 64 + kc * 32 + klo);
            a2.q2[1] = *(const u32x4*)(Ps + (m0 + rlo) * 64 + kc * 32 + 16 + klo);
#pragma unroll
            for (int s = 0; s < 16; ++s)
                b2.e[s] = vs[(kc * 32 + (s < 8 ? klo + s : 8 + klo + s)) * 32 + jt * 16 + rlo];
            c.v = __builtin_amdgcn_wmma_f32_16x16x32_bf16(false, a2.v, false, b2.v,
                                                          (short)0, c.v, false, false);
        }
#pragma unroll
        for (int r = 0; r < 8; ++r) {
            int row = m0 + rofs + r;
            if (row < Ntok)
                out[((size_t)win * Ntok + row) * Cdim + head * HD + jt * 16 + rlo] =
                    (bf16)c.e[r];
        }
    }
}

// =====================================================================
// LayerNorm2: xs (M,256) f32 -> xn2 (M,256) bf16. One wave32 per row.
// =====================================================================
__global__ __launch_bounds__(256) void ln2_kernel(
    const float* __restrict__ xs, const float* __restrict__ nw,
    const float* __restrict__ nb, bf16* __restrict__ xn)
{
    int wave = threadIdx.x >> 5, lane = threadIdx.x & 31;
    size_t row = (size_t)blockIdx.x * 8 + wave;
    const float* p = xs + row * Cdim;
    float vals[8], s = 0.f, sq = 0.f;
#pragma unroll
    for (int i = 0; i < 8; ++i) {
        float v = p[i * 32 + lane];
        vals[i] = v; s += v; sq += v * v;
    }
#pragma unroll
    for (int off = 16; off > 0; off >>= 1) {
        s  += __shfl_xor(s, off, 32);
        sq += __shfl_xor(sq, off, 32);
    }
    float mu = s * (1.f / Cdim);
    float rstd = rsqrtf(sq * (1.f / Cdim) - mu * mu + 1e-5f);
    bf16* o = xn + row * Cdim;
#pragma unroll
    for (int i = 0; i < 8; ++i) {
        int c = i * 32 + lane;
        o[c] = (bf16)((vals[i] - mu) * rstd * nw[c] + nb[c]);
    }
}

// small weight conversion kernel (f32 -> bf16)
__global__ __launch_bounds__(256) void cvt_f32_bf16_kernel(
    const float* __restrict__ a, bf16* __restrict__ b, int n)
{
    int i = blockIdx.x * blockDim.x + threadIdx.x;
    if (i < n) b[i] = (bf16)a[i];
}

// =====================================================================
extern "C" void kernel_launch(void* const* d_in, const int* in_sizes, int n_in,
                              void* d_out, int out_size, void* d_ws, size_t ws_size,
                              hipStream_t stream)
{
    const float* x       = (const float*)d_in[0];
    const float* qkv_wf  = (const float*)d_in[1];
    const float* q_bias  = (const float*)d_in[2];
    const float* v_bias  = (const float*)d_in[3];
    const float* proj_wf = (const float*)d_in[4];
    const float* proj_b  = (const float*)d_in[5];
    const float* rpb     = (const float*)d_in[6];
    const float* n1w     = (const float*)d_in[7];
    const float* n1b     = (const float*)d_in[8];
    const float* n2w     = (const float*)d_in[9];
    const float* n2b     = (const float*)d_in[10];
    const float* fc1_wf  = (const float*)d_in[11];
    const float* fc1_b   = (const float*)d_in[12];
    const float* fc2_wf  = (const float*)d_in[13];
    const float* fc2_b   = (const float*)d_in[14];

    char* ws = (char*)d_ws;
    size_t off = 0;
    float* xs  = (float*)(ws + off); off += (size_t)Mrows * Cdim * 4;
    bf16* xw   = (bf16*)(ws + off);  off += (size_t)Mrows * Cdim * 2;
    bf16* qkv  = (bf16*)(ws + off);  off += (size_t)3 * Mrows * Cdim * 2;
    bf16* ao   = (bf16*)(ws + off);  off += (size_t)Mrows * Cdim * 2;
    bf16* xn2  = (bf16*)(ws + off);  off += (size_t)Mrows * Cdim * 2;
    bf16* h1   = (bf16*)(ws + off);  off += (size_t)Mrows * 1024 * 2;
    bf16* wqkvb  = (bf16*)(ws + off); off += (size_t)768 * Cdim * 2;
    bf16* wprojb = (bf16*)(ws + off); off += (size_t)Cdim * Cdim * 2;
    bf16* wfc1b  = (bf16*)(ws + off); off += (size_t)1024 * Cdim * 2;
    bf16* wfc2b  = (bf16*)(ws + off); off += (size_t)Cdim * 1024 * 2;

    // weight conversion f32 -> bf16 (small, deterministic every call)
    cvt_f32_bf16_kernel<<<(768 * Cdim + 255) / 256, 256, 0, stream>>>(qkv_wf, wqkvb, 768 * Cdim);
    cvt_f32_bf16_kernel<<<(Cdim * Cdim + 255) / 256, 256, 0, stream>>>(proj_wf, wprojb, Cdim * Cdim);
    cvt_f32_bf16_kernel<<<(1024 * Cdim + 255) / 256, 256, 0, stream>>>(fc1_wf, wfc1b, 1024 * Cdim);
    cvt_f32_bf16_kernel<<<(Cdim * 1024 + 255) / 256, 256, 0, stream>>>(fc2_wf, wfc2b, Cdim * 1024);

    // 1. LN1 + shift + window partition
    ln1_part_kernel<<<Bsz * Hdim, 64, 0, stream>>>(x, n1w, n1b, xs, xw);

    // 2. QKV GEMM (M=100352, N=768, K=256)
    gemm_bf16_kernel<0><<<dim3(Mrows / 128, 768 / 64), 256, 0, stream>>>(
        xw, wqkvb, Cdim, q_bias, v_bias, qkv, nullptr);

    // 3. window attention
    attn_kernel<<<NWIN * NHead, 128, 0, stream>>>(qkv, rpb, ao);

    // 4. proj GEMM + window-reverse + unshift + residual (N=256, K=256)
    gemm_bf16_kernel<1><<<dim3(Mrows / 128, Cdim / 64), 256, 0, stream>>>(
        ao, wprojb, Cdim, proj_b, nullptr, nullptr, xs);

    // 5. LN2
    ln2_kernel<<<Mrows / 8, 256, 0, stream>>>(xs, n2w, n2b, xn2);

    // 6. fc1 + relu (N=1024, K=256)
    gemm_bf16_kernel<2><<<dim3(Mrows / 128, 1024 / 64), 256, 0, stream>>>(
        xn2, wfc1b, Cdim, fc1_b, nullptr, h1, nullptr);

    // 7. fc2 + residual + NCHW output (N=256, K=1024)
    gemm_bf16_kernel<3><<<dim3(Mrows / 128, Cdim / 64), 256, 0, stream>>>(
        h1, wfc2b, 1024, fc2_b, xs, nullptr, (float*)d_out);
}
